// VanillaRNN_46720654246152
// MI455X (gfx1250) — compile-verified, matrix operand order checked by reading, not run on hardware
//
#include <hip/hip_runtime.h>
#include <hip/hip_bf16.h>
#include <math.h>

#define BB 256   // batch
#define TT 128   // timesteps
#define HH 2048  // hidden
#define CC 10    // classes

typedef __attribute__((ext_vector_type(8)))  __bf16 v8bf;
typedef __attribute__((ext_vector_type(16))) __bf16 v16bf;
typedef __attribute__((ext_vector_type(8)))  float  v8f;
typedef __attribute__((ext_vector_type(4)))  float  v4f;

// ---------------------------------------------------------------------------
// W (fp32, row-major HxH) -> bf16, vectorized x4
// ---------------------------------------------------------------------------
__global__ __launch_bounds__(256) void convert_w_bf16(const float* __restrict__ src,
                                                      __bf16* __restrict__ dst) {
    int i = (blockIdx.x * blockDim.x + threadIdx.x) * 4;
    v4f v = *(const v4f*)(src + i);
    dst[i + 0] = (__bf16)v.x;
    dst[i + 1] = (__bf16)v.y;
    dst[i + 2] = (__bf16)v.z;
    dst[i + 3] = (__bf16)v.w;
}

// ---------------------------------------------------------------------------
// Step 0: h = tanh(x[:,0] * U^T + bh)   (h_prev == 0, so no GEMM)
// ---------------------------------------------------------------------------
__global__ __launch_bounds__(256) void rnn_first_step(const float* __restrict__ x,
                                                      const float* __restrict__ U,
                                                      const float* __restrict__ bh,
                                                      __bf16* __restrict__ h) {
    int idx = blockIdx.x * blockDim.x + threadIdx.x;   // [0, BB*HH)
    int b = idx >> 11;           // / HH
    int i = idx & (HH - 1);      // % HH
    float v = x[b * TT + 0] * U[i] + bh[i];
    h[idx] = (__bf16)tanhf(v);
}

// ---------------------------------------------------------------------------
// Step t (t>=1): hnext = tanh( hprev @ W^T + x[:,t]*U^T + bh )
// M=BB, N=HH, K=HH.  A = hprev (bf16 row-major), B[k,n] = W[n,k].
//
// Block: 256 threads = 8 waves arranged 2(M) x 4(N); macro-tile 64(M) x 128(N).
// Each wave: 32x32 output = 2x2 WMMA tiles (4 f32 accumulators).
// K-slices of 32 staged through a 3-deep LDS ring via async global->LDS;
// s_wait_asynccnt 3 retires exactly the oldest stage (in-order completion),
// so each transfer gets a full iteration of latency cover.
// Grid: (BB/64) * (HH/128) = 4 * 16 = 64 blocks.
// ---------------------------------------------------------------------------
__global__ __launch_bounds__(256) void rnn_step(const __bf16* __restrict__ hprev,
                                                const __bf16* __restrict__ Wb,
                                                const float*  __restrict__ x,
                                                const float*  __restrict__ U,
                                                const float*  __restrict__ bh,
                                                __bf16* __restrict__ hnext,
                                                int t) {
    __shared__ __bf16 As[3][64 * 32];    // 4 KB per buffer
    __shared__ __bf16 Bs[3][128 * 32];   // 8 KB per buffer

    const int bn = blockIdx.x & 15;                 // 16 N-blocks
    const int bm = blockIdx.x >> 4;                 // 4  M-blocks
    const int m0 = bm * 64;
    const int n0 = bn * 128;

    const int tid  = threadIdx.x;
    const int lane = tid & 31;
    const int wave = tid >> 5;
    const int wm = wave >> 2;                       // 0..1 -> 32 M-rows
    const int wn = wave & 3;                        // 0..3 -> 32 N-cols

    // 16-bit fragment layout (ISA 7.12.2): lanes 0-15: row=lane, K in {0..7,16..23};
    // lanes 16-31: row=lane-16, K in {8..15,24..31}. Contiguous 16B x2 per lane.
    const int r    = lane & 15;
    const int koff = (lane >> 4) * 8;               // bf16 elements

    // --- staging assignment: 256 threads move 12 KB (A:64x32, B:128x32) ---
    const int srow = tid >> 2;                      // 0..63
    const int sc   = (tid & 3) * 8;                 // 0,8,16,24 (bf16 elems)

    const __bf16* gA  = hprev + (size_t)(m0 + srow)      * HH + sc;
    const __bf16* gB0 = Wb    + (size_t)(n0 + srow)      * HH + sc;
    const __bf16* gB1 = Wb    + (size_t)(n0 + 64 + srow) * HH + sc;

    const unsigned ldsA  = (unsigned)(uintptr_t)&As[0][srow * 32 + sc];
    const unsigned ldsB0 = (unsigned)(uintptr_t)&Bs[0][srow * 32 + sc];
    const unsigned ldsB1 = (unsigned)(uintptr_t)&Bs[0][(64 + srow) * 32 + sc];
    const unsigned strideA = 64 * 32 * 2;           // bytes between A buffers
    const unsigned strideB = 128 * 32 * 2;          // bytes between B buffers

    // stage K-slice `slice` (k = slice*32) into ring buffer slice%3
    auto stage = [&](int buf, int slice) {
        int k = slice * 32;
        unsigned la  = ldsA  + (unsigned)buf * strideA;
        unsigned lb0 = ldsB0 + (unsigned)buf * strideB;
        unsigned lb1 = ldsB1 + (unsigned)buf * strideB;
        unsigned long long a  = (unsigned long long)(uintptr_t)(gA  + k);
        unsigned long long b0 = (unsigned long long)(uintptr_t)(gB0 + k);
        unsigned long long b1 = (unsigned long long)(uintptr_t)(gB1 + k);
        asm volatile("global_load_async_to_lds_b128 %0, %1, off\n\t"
                     "global_load_async_to_lds_b128 %2, %3, off\n\t"
                     "global_load_async_to_lds_b128 %4, %5, off"
                     :: "v"(la), "v"(a), "v"(lb0), "v"(b0), "v"(lb1), "v"(b1)
                     : "memory");
    };

    auto loadFrag = [&](const __bf16* p) -> v16bf {
        v8bf lo = *(const v8bf*)p;              // ds_load_b128
        v8bf hi = *(const v8bf*)(p + 16);       // ds_load_b128
        return __builtin_shufflevector(lo, hi,
                   0,1,2,3,4,5,6,7,8,9,10,11,12,13,14,15);
    };

    v8f acc[2][2] = {};

    auto compute = [&](int buf) {
        const __bf16* Ab = As[buf];
        const __bf16* Bb = Bs[buf];
        v16bf a0 = loadFrag(&Ab[(wm * 32 +      r) * 32 + koff]);
        v16bf a1 = loadFrag(&Ab[(wm * 32 + 16 + r) * 32 + koff]);
        v16bf b0 = loadFrag(&Bb[(wn * 32 +      r) * 32 + koff]);
        v16bf b1 = loadFrag(&Bb[(wn * 32 + 16 + r) * 32 + koff]);
        acc[0][0] = __builtin_amdgcn_wmma_f32_16x16x32_bf16(false, a0, false, b0,
                        (short)0, acc[0][0], false, false);
        acc[0][1] = __builtin_amdgcn_wmma_f32_16x16x32_bf16(false, a0, false, b1,
                        (short)0, acc[0][1], false, false);
        acc[1][0] = __builtin_amdgcn_wmma_f32_16x16x32_bf16(false, a1, false, b0,
                        (short)0, acc[1][0], false, false);
        acc[1][1] = __builtin_amdgcn_wmma_f32_16x16x32_bf16(false, a1, false, b1,
                        (short)0, acc[1][1], false, false);
    };

    const int NSLICE = HH / 32;                 // 64 K-slices

    // ---- prologue: 2 stages in flight ----
    stage(0, 0);
    stage(1, 1);

    int cb = 0;                                  // buffer holding slice i
    #pragma unroll 1
    for (int i = 0; i < NSLICE - 2; ++i) {
        // oldest stage (slice i, 3 asyncs) retired; stage for slice i+1 may remain
        asm volatile("s_wait_asynccnt 3" ::: "memory");
        __syncthreads();                         // slice i visible; buffer (i+2)%3 free
        int sb = cb + 2; if (sb >= 3) sb -= 3;
        stage(sb, i + 2);
        compute(cb);
        if (++cb == 3) cb = 0;
    }
    // tail: slice NSLICE-2 (one stage still in flight behind it)
    asm volatile("s_wait_asynccnt 3" ::: "memory");
    __syncthreads();
    compute(cb);
    if (++cb == 3) cb = 0;
    // tail: slice NSLICE-1 (drain)
    asm volatile("s_wait_asynccnt 0" ::: "memory");
    __syncthreads();
    compute(cb);

    // ---- fused epilogue: + x[:,t]*U^T + bh, tanh, store bf16 ----
    const int nloc = lane & 15;
    const int moff = (lane >> 4) * 8;
    #pragma unroll
    for (int mt = 0; mt < 2; ++mt) {
        #pragma unroll
        for (int nt = 0; nt < 2; ++nt) {
            int n = n0 + wn * 32 + nt * 16 + nloc;
            float un  = U[n];
            float bhn = bh[n];
            #pragma unroll
            for (int v = 0; v < 8; ++v) {
                int m = m0 + wm * 32 + mt * 16 + moff + v;
                float val = acc[mt][nt][v] + x[m * TT + t] * un + bhn;
                hnext[(size_t)m * HH + n] = (__bf16)tanhf(val);
            }
        }
    }
}

// ---------------------------------------------------------------------------
// Final projection: p[b,c] = h_last[b,:] . V[c,:] + bp[c]   (256x10, tiny)
// ---------------------------------------------------------------------------
__global__ __launch_bounds__(256) void rnn_proj(const __bf16* __restrict__ h,
                                                const float* __restrict__ V,
                                                const float* __restrict__ bp,
                                                float* __restrict__ out) {
    int idx = blockIdx.x * blockDim.x + threadIdx.x;
    if (idx >= BB * CC) return;
    int b = idx / CC;
    int c = idx % CC;
    const __bf16* hr = h + (size_t)b * HH;
    const float*  vr = V + (size_t)c * HH;
    float acc = bp[c];
    #pragma unroll 8
    for (int k = 0; k < HH; ++k)
        acc += (float)hr[k] * vr[k];
    out[idx] = acc;
}

// ---------------------------------------------------------------------------
extern "C" void kernel_launch(void* const* d_in, const int* in_sizes, int n_in,
                              void* d_out, int out_size, void* d_ws, size_t ws_size,
                              hipStream_t stream) {
    const float* x  = (const float*)d_in[0];   // [B, T]
    const float* U  = (const float*)d_in[1];   // [H, 1]
    const float* W  = (const float*)d_in[2];   // [H, H]
    const float* V  = (const float*)d_in[3];   // [C, H]
    const float* bh = (const float*)d_in[4];   // [H]
    const float* bp = (const float*)d_in[5];   // [C]
    float* out = (float*)d_out;                // [B, C]

    // Workspace layout: Wbf16 (8MB) | h ping (1MB) | h pong (1MB)
    char*   ws = (char*)d_ws;
    __bf16* Wb = (__bf16*)ws;
    __bf16* h0 = (__bf16*)(ws + (size_t)HH * HH * sizeof(__bf16));
    __bf16* h1 = h0 + (size_t)BB * HH;

    // 1) W -> bf16 (stays L2-resident: 8MB << 192MB)
    convert_w_bf16<<<(HH * HH / 4) / 256, 256, 0, stream>>>(W, Wb);

    // 2) step 0 (h_prev == 0): pure elementwise
    rnn_first_step<<<(BB * HH) / 256, 256, 0, stream>>>(x, U, bh, h0);

    // 3) steps 1..T-1: chained bf16 WMMA GEMMs, ping-pong h
    __bf16* cur = h0;
    __bf16* nxt = h1;
    for (int t = 1; t < TT; ++t) {
        rnn_step<<<(BB / 64) * (HH / 128), 256, 0, stream>>>(cur, Wb, x, U, bh, nxt, t);
        __bf16* tmp = cur; cur = nxt; nxt = tmp;
    }

    // 4) final projection from last hidden state
    rnn_proj<<<(BB * CC + 255) / 256, 256, 0, stream>>>(cur, V, bp, out);
}
